// MyLSTM_52621939310798
// MI455X (gfx1250) — compile-verified, matrix operand order checked by reading, not run on hardware
//
#include <hip/hip_runtime.h>
#include <hip/hip_bf16.h>

// ---------------------------------------------------------------------------
// LSTM forward for MI455X (gfx1250, wave32, WMMA bf16).
//
// Phase 1: convert input / W_i2h / W_h2h / h0 to bf16 (workspace), init state.
// Phase 2: x_comp GEMM. Per block: A-tile (64 rows x 1024 k, 128KB) staged in
//          LDS; fully-unrolled k-loop, scheduler-pipelined wmma bf16.
// Phase 3: persistent scan (32 resident blocks). Per step: h staged into LDS
//          via async global->LDS copies (ASYNCcnt), gates = x_comp[t] +
//          h @ W_h2h^T + b_h2h, grid barrier -> pointwise update -> barrier.
// ---------------------------------------------------------------------------

namespace {
constexpr int kT = 512;
constexpr int kB = 64;
constexpr int kI = 1024;
constexpr int kH = 1024;
constexpr int kG = 4096;                 // 4 * kH
constexpr int kKSteps = 32;              // 1024 / 32
constexpr int kScanBlocks = 32;
constexpr int kScanThreads = kScanBlocks * 256;  // 8192
constexpr int kLdsStride = kH + 8;       // +8 bf16 pad: 2064B row stride,
                                         // 16B aligned, 4-bank rotation
}

typedef __attribute__((ext_vector_type(16))) __bf16 v16bf;
typedef __attribute__((ext_vector_type(8)))  __bf16 v8bf;
typedef __attribute__((ext_vector_type(8)))  float  v8f;

// Per ISA 16-bit A/B layout: lanes 0-15 hold K = {kb*32+0..7, kb*32+16..23},
// lanes 16-31 hold K = {kb*32+8..15, kb*32+24..31}. The lane's half*8 column
// shift is folded into the base pointer; offsets stay immediates.
__device__ __forceinline__ v16bf load_frag_bf16(const __bf16* base, int koff) {
  v8bf lo = *reinterpret_cast<const v8bf*>(base + koff);
  v8bf hi = *reinterpret_cast<const v8bf*>(base + koff + 16);
  return __builtin_shufflevector(lo, hi, 0, 1, 2, 3, 4, 5, 6, 7,
                                 8, 9, 10, 11, 12, 13, 14, 15);
}

__device__ __forceinline__ float sigmoid_f(float x) {
  return 1.0f / (1.0f + __expf(-x));
}

__device__ __forceinline__ void wait_asynccnt0() {
#if __has_builtin(__builtin_amdgcn_s_wait_asynccnt)
  __builtin_amdgcn_s_wait_asynccnt(0);
#else
  asm volatile("s_wait_asynccnt 0x0" ::: "memory");
#endif
}

// Fully-unrolled 4xM-tile x 1xN-tile WMMA k-loop. abLo points at the lane's
// LDS row (lidx) with half*8 folded in; M-tile strides are compile-time
// constants so mt0/mt1 become DS-offset immediates and mt2/mt3 share one
// CSE'd base. wbase is the lane's global W row (half*8 folded in).
__device__ __forceinline__ void gemm_kloop(const __bf16* abLo,
                                           const __bf16* wbase, v8f acc[4]) {
#pragma unroll
  for (int kb = 0; kb < kKSteps; ++kb) {
    const int koff = kb * 32;
    v16bf bfrag = load_frag_bf16(wbase, koff);
    v16bf a0 = load_frag_bf16(abLo, koff);
    v16bf a1 = load_frag_bf16(abLo + 16 * kLdsStride, koff);
    v16bf a2 = load_frag_bf16(abLo + 32 * kLdsStride, koff);
    v16bf a3 = load_frag_bf16(abLo + 48 * kLdsStride, koff);
    acc[0] = __builtin_amdgcn_wmma_f32_16x16x32_bf16(
        false, a0, false, bfrag, (short)0, acc[0], false, false);
    acc[1] = __builtin_amdgcn_wmma_f32_16x16x32_bf16(
        false, a1, false, bfrag, (short)0, acc[1], false, false);
    acc[2] = __builtin_amdgcn_wmma_f32_16x16x32_bf16(
        false, a2, false, bfrag, (short)0, acc[2], false, false);
    acc[3] = __builtin_amdgcn_wmma_f32_16x16x32_bf16(
        false, a3, false, bfrag, (short)0, acc[3], false, false);
  }
}

// ---------------------------------------------------------------------------
// Phase 1: f32 -> bf16 conversion + state init + barrier reset
// ---------------------------------------------------------------------------
__global__ __launch_bounds__(256) void lstm_convert_init(
    const float* __restrict__ input, const float* __restrict__ h0,
    const float* __restrict__ c0, const float* __restrict__ Wi,
    const float* __restrict__ Wh,
    __bf16* __restrict__ in_bf, __bf16* __restrict__ Wi_bf,
    __bf16* __restrict__ Wh_bf,
    float* __restrict__ h_f, float* __restrict__ c_f,
    __bf16* __restrict__ h_bf, unsigned* __restrict__ bar) {
  const size_t tid = (size_t)blockIdx.x * blockDim.x + threadIdx.x;
  const size_t nth = (size_t)gridDim.x * blockDim.x;
  const size_t n_in = (size_t)kT * kB * kI;
  for (size_t i = tid; i < n_in; i += nth) in_bf[i] = (__bf16)input[i];
  const size_t n_w = (size_t)kG * kI;
  for (size_t i = tid; i < n_w; i += nth) Wi_bf[i] = (__bf16)Wi[i];
  for (size_t i = tid; i < n_w; i += nth) Wh_bf[i] = (__bf16)Wh[i];
  const size_t n_h = (size_t)kB * kH;
  for (size_t i = tid; i < n_h; i += nth) {
    float hv = h0[i];
    h_f[i] = hv;
    h_bf[i] = (__bf16)hv;
    c_f[i] = c0[i];
  }
  if (tid == 0) { bar[0] = 0u; bar[1] = 0u; }
}

// ---------------------------------------------------------------------------
// Phase 2: x_comp GEMM  (M = T*B = 32768, N = 4096, K = 1024)
// ---------------------------------------------------------------------------
__global__ __launch_bounds__(256) void lstm_gemm_i2h(
    const __bf16* __restrict__ A,      // (T*B) x kI   row-major
    const __bf16* __restrict__ W,      // kG x kI      row-major (= B col-major)
    const float* __restrict__ bias,    // kG
    float* __restrict__ out) {         // (T*B) x kG
  __shared__ __align__(16) __bf16 lds_a[64 * kLdsStride];

  const int wave = threadIdx.x >> 5;
  const int lane = threadIdx.x & 31;
  const int half = lane >> 4;
  const int lidx = lane & 15;
  const int mgroup = blockIdx.x >> 5;        // 512 groups of 4 M-tiles
  const int ngroup = blockIdx.x & 31;        // 32 groups of 8 N-tiles
  const int ntile = ngroup * 8 + wave;       // 0..255
  const int m0 = mgroup * 64;
  const int n = ntile * 16 + lidx;

  // Stage A tile (64 x 1024 bf16) into LDS: global_load_b128 + ds_store_b128.
  {
    const __bf16* __restrict__ base = A + (size_t)m0 * kI;
#pragma unroll
    for (int it = 0; it < 32; ++it) {
      const int c = threadIdx.x + it * 256;   // 16B chunk id (8192 total)
      const int row = c >> 7;                 // 128 chunks per row
      const int col = (c & 127) * 8;
      *reinterpret_cast<v8bf*>(lds_a + row * kLdsStride + col) =
          *reinterpret_cast<const v8bf*>(base + (size_t)row * kI + col);
    }
  }
  __syncthreads();

  const __bf16* abLo = lds_a + (size_t)lidx * kLdsStride + half * 8;
  const __bf16* wbase = W + (size_t)n * kI + half * 8;

  v8f acc[4] = {};
  gemm_kloop(abLo, wbase, acc);

  const float bv = bias[n];
#pragma unroll
  for (int mt = 0; mt < 4; ++mt) {
#pragma unroll
    for (int vg = 0; vg < 8; ++vg) {
      const int m = m0 + mt * 16 + half * 8 + vg;   // C layout: M = vg + 8*half
      out[(size_t)m * kG + n] = acc[mt][vg] + bv;
    }
  }
}

// ---------------------------------------------------------------------------
// Grid barrier (sense-reversal on global atomics). All 32 blocks resident.
// ---------------------------------------------------------------------------
__device__ __forceinline__ void grid_barrier(unsigned* __restrict__ bar) {
  __threadfence();
  __syncthreads();
  if (threadIdx.x == 0) {
    unsigned gen = __hip_atomic_load(&bar[1], __ATOMIC_RELAXED,
                                     __HIP_MEMORY_SCOPE_AGENT);
    unsigned arrived = __hip_atomic_fetch_add(&bar[0], 1u, __ATOMIC_ACQ_REL,
                                              __HIP_MEMORY_SCOPE_AGENT);
    if (arrived == (unsigned)(kScanBlocks - 1)) {
      __hip_atomic_store(&bar[0], 0u, __ATOMIC_RELAXED,
                         __HIP_MEMORY_SCOPE_AGENT);
      __hip_atomic_store(&bar[1], gen + 1u, __ATOMIC_RELEASE,
                         __HIP_MEMORY_SCOPE_AGENT);
    } else {
      while (__hip_atomic_load(&bar[1], __ATOMIC_ACQUIRE,
                               __HIP_MEMORY_SCOPE_AGENT) == gen) {
        __builtin_amdgcn_s_sleep(1);
      }
    }
  }
  __syncthreads();
}

// ---------------------------------------------------------------------------
// Phase 3: persistent recurrent scan.
// ---------------------------------------------------------------------------
__global__ __launch_bounds__(256) void lstm_scan(
    const __bf16* __restrict__ Wh,     // kG x kH bf16 (row-major)
    const float* __restrict__ xcomp,   // kT x kB x kG f32
    const float* __restrict__ bh,      // kG
    const int* __restrict__ lengths,   // kB
    float* __restrict__ h_f, float* __restrict__ c_f,
    __bf16* __restrict__ h_bf,
    float* __restrict__ gates,         // kB x kG f32 scratch
    float* __restrict__ out_hs,        // kT x kB x kH
    float* __restrict__ out_h,         // kB x kH
    float* __restrict__ out_c,         // kB x kH
    unsigned* __restrict__ bar) {
  __shared__ __align__(16) __bf16 lds_h[64 * kLdsStride];

  const int wave = threadIdx.x >> 5;
  const int lane = threadIdx.x & 31;
  const int half = lane >> 4;
  const int lidx = lane & 15;
  const int ntile = blockIdx.x * 8 + wave;   // 0..255
  const int n = ntile * 16 + lidx;
  const int gtid = blockIdx.x * 256 + threadIdx.x;  // 0..8191

  const __bf16* abLo = lds_h + (size_t)lidx * kLdsStride + half * 8;
  const __bf16* wbase = Wh + (size_t)n * kH + half * 8;
  const float bv = bh[n];

  for (int t = 0; t < kT; ++t) {
    // ---- stage h into LDS with async global->LDS copies ----
    // lds_h is the only LDS object in this kernel -> allocated at offset 0.
#pragma unroll
    for (int it = 0; it < 32; ++it) {
      const int c = threadIdx.x + it * 256;   // 16B chunk id (8192 total)
      const int row = c >> 7;
      const int col = (c & 127) * 8;
      const unsigned ldsoff = (unsigned)((row * kLdsStride + col) * 2);
      const __bf16* g = h_bf + (size_t)row * kH + col;
      asm volatile("global_load_async_to_lds_b128 %0, %1, off"
                   :
                   : "v"(ldsoff), "v"(g)
                   : "memory");
    }
    if (t + 1 < kT) {  // prefetch next step's x_comp slice into cache
      __builtin_prefetch(xcomp + (size_t)(t + 1) * kB * kG + (size_t)gtid * 32,
                         0, 1);
    }
    wait_asynccnt0();
    __syncthreads();

    // ---- GEMM phase: gates = x_comp[t] + h @ Wh^T + b_h2h ----
    const float* __restrict__ xr = xcomp + (size_t)t * kB * kG;
    v8f acc[4];
#pragma unroll
    for (int mt = 0; mt < 4; ++mt) {
#pragma unroll
      for (int vg = 0; vg < 8; ++vg) {
        const int m = mt * 16 + half * 8 + vg;
        acc[mt][vg] = xr[(size_t)m * kG + n] + bv;
      }
    }
    gemm_kloop(abLo, wbase, acc);
#pragma unroll
    for (int mt = 0; mt < 4; ++mt) {
#pragma unroll
      for (int vg = 0; vg < 8; ++vg) {
        const int m = mt * 16 + half * 8 + vg;
        gates[(size_t)m * kG + n] = acc[mt][vg];
      }
    }
    grid_barrier(bar);

    // ---- pointwise LSTM cell update (gate order: f, i, o, cand) ----
    for (int e = gtid; e < kB * kH; e += kScanThreads) {
      const int b = e >> 10;           // e / kH
      const int j = e & (kH - 1);      // e % kH
      const float* __restrict__ g = gates + (size_t)b * kG;
      const float fg = sigmoid_f(g[j]);
      const float ig = sigmoid_f(g[j + kH]);
      const float og = sigmoid_f(g[j + 2 * kH]);
      const float cd = tanhf(g[j + 3 * kH]);
      const float co = c_f[e];
      const float cn = fg * co + ig * cd;
      const float hn = og * tanhf(co);   // reference uses tanh of OLD c
      float hcur;
      if (t < lengths[b]) {
        h_f[e] = hn;
        c_f[e] = cn;
        hcur = hn;
      } else {
        hcur = h_f[e];
      }
      out_hs[(size_t)t * kB * kH + e] = hcur;
      h_bf[e] = (__bf16)hcur;
    }
    grid_barrier(bar);   // h_bf ready / gates + LDS reusable next step
  }

  // final h, c
  for (int e = gtid; e < kB * kH; e += kScanThreads) {
    out_h[e] = h_f[e];
    out_c[e] = c_f[e];
  }
}

// ---------------------------------------------------------------------------
extern "C" void kernel_launch(void* const* d_in, const int* in_sizes, int n_in,
                              void* d_out, int out_size, void* d_ws,
                              size_t ws_size, hipStream_t stream) {
  (void)in_sizes; (void)n_in; (void)out_size; (void)ws_size;
  const float* input = (const float*)d_in[0];
  const float* h0    = (const float*)d_in[1];
  const float* c0    = (const float*)d_in[2];
  const float* Wi    = (const float*)d_in[3];
  const float* bi    = (const float*)d_in[4];
  const float* Wh    = (const float*)d_in[5];
  const float* bh    = (const float*)d_in[6];
  const int*   len   = (const int*)d_in[7];
  float* out = (float*)d_out;

  char* ws = (char*)d_ws;
  size_t off = 0;
  auto take = [&](size_t bytes) -> char* {
    char* p = ws + off;
    off = (off + bytes + 255) & ~(size_t)255;
    return p;
  };
  __bf16* in_bf = (__bf16*)take((size_t)kT * kB * kI * 2);   //  64 MB
  __bf16* Wi_bf = (__bf16*)take((size_t)kG * kI * 2);        //   8 MB
  __bf16* Wh_bf = (__bf16*)take((size_t)kG * kH * 2);        //   8 MB
  float*  xcomp = (float*)take((size_t)kT * kB * kG * 4);    // 512 MB
  float*  h_f   = (float*)take((size_t)kB * kH * 4);
  float*  c_f   = (float*)take((size_t)kB * kH * 4);
  __bf16* h_bf  = (__bf16*)take((size_t)kB * kH * 2);
  float*  gates = (float*)take((size_t)kB * kG * 4);         //   1 MB
  unsigned* bar = (unsigned*)take(256);

  float* out_hs = out;
  float* out_h  = out + (size_t)kT * kB * kH;
  float* out_c  = out_h + (size_t)kB * kH;

  // Phase 1: convert + init
  lstm_convert_init<<<2048, 256, 0, stream>>>(input, h0, c0, Wi, Wh, in_bf,
                                              Wi_bf, Wh_bf, h_f, c_f, h_bf,
                                              bar);
  // Phase 2: input projection GEMM (M=32768, N=4096, K=1024)
  lstm_gemm_i2h<<<512 * 32, 256, 0, stream>>>(in_bf, Wi_bf, bi, xcomp);
  // Phase 3: persistent recurrent scan (32 resident blocks, grid barriers)
  lstm_scan<<<kScanBlocks, 256, 0, stream>>>(Wh_bf, xcomp, bh, len, h_f, c_f,
                                             h_bf, gates, out_hs, out_h,
                                             out_c, bar);
}
